// EEG_SequentialCompressionUnit_28355374088685
// MI455X (gfx1250) — matrix operand order repaired and checked by reading, not compiled, not run on hardware
//
#include <hip/hip_runtime.h>
#include <hip/hip_bf16.h>

// ---------------------------------------------------------------------------
// WTA-LIF scan, B=2048, C=256, T=512.
//  K1: bf16 WMMA GEMM  Y[t,b,c] = sum_k x[b,k,t] * Wf[c,k]
//      - A tile (16 rows x 256 k) staged ONCE per workgroup into 8 KB LDS
//        (one global load per x element total), WMMA A-frags fed by
//        ds_load_b128 pairs.  B-frags direct from L2-hot Wf (each element
//        read exactly once per WG; lane holds one contiguous 16-K run per
//        the k order also used for the A-frags).
//  K2: sequential scan, one wave32 per batch row; Y rows double-buffered in
//      LDS via GLOBAL_LOAD_ASYNC_TO_LDS_B128 + s_wait_asynccnt (per-wave
//      ASYNCcnt, in-order completion).  Recurrent term = 1-row gather of WrT
//      (spike_out is one-hot).  Output = exp(pre-reset v) at t=T-1.
// Roofline: ~3 GiB HBM traffic -> ~130 us at 23.3 TB/s; 137 GFLOP of WMMA is
// negligible against that.
// ---------------------------------------------------------------------------

#define Bsz 2048
#define Csz 256
#define Tsz 512
#define DECAY 0.83333333333333333f   // 1 - 1/6
#define VTH 3.0f

typedef __attribute__((ext_vector_type(16))) __bf16 bf16x16;
typedef __attribute__((ext_vector_type(8)))  __bf16 bf16x8;
typedef __attribute__((ext_vector_type(8)))  float  f32x8;

// ---------------- K0: WrT[k*C + c] = Wr[c*C + k] ---------------------------
__global__ void wr_transpose_kernel(const float* __restrict__ Wr,
                                    float* __restrict__ WrT) {
    int idx = blockIdx.x * blockDim.x + threadIdx.x;   // 0 .. C*C-1
    int k = idx >> 8;
    int c = idx & 255;
    WrT[idx] = Wr[c * Csz + k];
}

// ---------------- K1: Y[t,b,c] via v_wmma_f32_16x16x32_bf16 ----------------
// grid = (T, B/16); block = 256 threads = 8 waves.
// Wave w computes n-tiles (2w)*16 and (2w+1)*16 for the 16-row b block.
__global__ void xwf_gemm_kernel(const float* __restrict__ x,
                                const float* __restrict__ Wf,
                                float* __restrict__ Y) {
    __shared__ __align__(16) __bf16 Alds[16 * 256];    // 8 KB: A tile, [m][k]

    const int t    = blockIdx.x;
    const int b0   = blockIdx.y << 4;
    const int tid  = threadIdx.x;
    const int wave = tid >> 5;
    const int lane = tid & 31;

    // ---- cooperative A-tile staging: thread -> (m = tid&15, 16 k values) --
    {
        const int m  = tid & 15;
        const int kg = tid >> 4;                       // 0..15
        const size_t base =
            (size_t)(b0 + m) * (Csz * Tsz) + (size_t)(kg * 16) * Tsz + t;
        float tmp[16];
        #pragma unroll
        for (int j = 0; j < 16; ++j)
            tmp[j] = x[base + (size_t)j * Tsz];        // strided gather, 1x per elem
        bf16x8 v0, v1;
        #pragma unroll
        for (int j = 0; j < 8; ++j) {
            v0[j] = (__bf16)tmp[j];
            v1[j] = (__bf16)tmp[8 + j];
        }
        bf16x8* dst = (bf16x8*)(Alds + m * 256 + kg * 16);
        dst[0] = v0;                                   // ds_store_b128
        dst[1] = v1;
    }
    __syncthreads();

    // A fragment layout: contiguous K split per lane half (matches B)
    const int khalf = (lane >> 4) << 4;                // 0 or 16
    // B fragment layout (same k order as A): contiguous K split per lane half
    const int bhalf = (lane >> 4) << 4;                // 0 or 16
    const __bf16* arow = Alds + (lane & 15) * 256;

    f32x8 acc0 = {}, acc1 = {};

    #pragma unroll
    for (int ks = 0; ks < Csz / 32; ++ks) {
        const int k0 = ks * 32;

        // ---- A fragment from LDS: two 16B runs -> ds_load_b128 x2 ----
        const bf16x8 alo = *(const bf16x8*)(arow + k0 + khalf);        // K khalf+0..7
        const bf16x8 ahi = *(const bf16x8*)(arow + k0 + 8 + khalf);    // K khalf+8..15
        const bf16x16 afrag = __builtin_shufflevector(
            alo, ahi, 0, 1, 2, 3, 4, 5, 6, 7, 8, 9, 10, 11, 12, 13, 14, 15);

        // ---- B fragments: B[k,n] = Wf[n,k]; lane holds K = bhalf+0..15 ----
        #pragma unroll
        for (int nt = 0; nt < 2; ++nt) {
            const int n0 = ((wave << 1) + nt) << 4;
            const int cc = n0 + (lane & 15);
            const float* wrow = Wf + (size_t)cc * Csz + k0 + bhalf;    // 64B run
            const float4 f0 = ((const float4*)wrow)[0];
            const float4 f1 = ((const float4*)wrow)[1];
            const float4 f2 = ((const float4*)wrow)[2];
            const float4 f3 = ((const float4*)wrow)[3];
            bf16x16 bfrag;
            bfrag[0]=(__bf16)f0.x;  bfrag[1]=(__bf16)f0.y;
            bfrag[2]=(__bf16)f0.z;  bfrag[3]=(__bf16)f0.w;
            bfrag[4]=(__bf16)f1.x;  bfrag[5]=(__bf16)f1.y;
            bfrag[6]=(__bf16)f1.z;  bfrag[7]=(__bf16)f1.w;
            bfrag[8]=(__bf16)f2.x;  bfrag[9]=(__bf16)f2.y;
            bfrag[10]=(__bf16)f2.z; bfrag[11]=(__bf16)f2.w;
            bfrag[12]=(__bf16)f3.x; bfrag[13]=(__bf16)f3.y;
            bfrag[14]=(__bf16)f3.z; bfrag[15]=(__bf16)f3.w;

            if (nt == 0)
                acc0 = __builtin_amdgcn_wmma_f32_16x16x32_bf16(
                           false, afrag, false, bfrag, (short)0, acc0, false, false);
            else
                acc1 = __builtin_amdgcn_wmma_f32_16x16x32_bf16(
                           false, afrag, false, bfrag, (short)0, acc1, false, false);
        }
    }

    // ---- store D (lanes 0-15: M=r, lanes 16-31: M=r+8; N = lane&15) ----
    const int nlo  = lane & 15;
    const int mofs = (lane >> 4) << 3;
    #pragma unroll
    for (int r = 0; r < 8; ++r) {
        const size_t row = (size_t)t * Bsz + (b0 + r + mofs);
        Y[row * Csz + (wave << 5) + nlo]      = acc0[r];
        Y[row * Csz + (wave << 5) + 16 + nlo] = acc1[r];
    }
}

// ---------------- K2: WTA-LIF scan, one wave32 per batch row ---------------
// block = 256 threads = 8 rows; grid = B/8.  Channel c = i*32 + lane.
// Y rows double-buffered in LDS via async global->LDS DMA (ASYNCcnt).
__global__ void wta_lif_scan_kernel(const float* __restrict__ Y,
                                    const float* __restrict__ WrT,
                                    float* __restrict__ out) {
    __shared__ __align__(16) float ybuf[8][2][Csz];    // 16 KB

    const int wave = threadIdx.x >> 5;
    const int lane = threadIdx.x & 31;
    const int b = (blockIdx.x << 3) + wave;

    // LDS byte offsets of this wave's two buffers (LDS aperture: addr[31:0])
    const unsigned lds0 = (unsigned)(size_t)(&ybuf[wave][0][0]) + lane * 16u;
    const unsigned lds1 = (unsigned)(size_t)(&ybuf[wave][1][0]) + lane * 16u;

    float v[8];
    #pragma unroll
    for (int i = 0; i < 8; ++i) v[i] = 0.0f;
    bool spiked = false;
    int  cstar  = 0;
    float pre[8];

    // issue async copy of row 0 into buffer 0 (2 x B128 per lane = 1 KB row)
    {
        const float* g = Y + (size_t)b * Csz + lane * 4;
        asm volatile("global_load_async_to_lds_b128 %0, %1, off"
                     :: "v"(lds0), "v"(g) : "memory");
        asm volatile("global_load_async_to_lds_b128 %0, %1, off"
                     :: "v"(lds0 + 512u), "v"(g + 128) : "memory");
    }

    for (int t = 0; t < Tsz; ++t) {
        if (t + 1 < Tsz) {
            // prefetch row t+1 into the other buffer while computing row t
            const unsigned ldsn = ((t + 1) & 1) ? lds1 : lds0;
            const float* g = Y + ((size_t)(t + 1) * Bsz + b) * Csz + lane * 4;
            asm volatile("global_load_async_to_lds_b128 %0, %1, off"
                         :: "v"(ldsn), "v"(g) : "memory");
            asm volatile("global_load_async_to_lds_b128 %0, %1, off"
                         :: "v"(ldsn + 512u), "v"(g + 128) : "memory");
            // async loads complete in order: <=2 outstanding => row t landed
            asm volatile("s_wait_asynccnt 2" ::: "memory");
        } else {
            asm volatile("s_wait_asynccnt 0" ::: "memory");
        }
        const float* yrow = &ybuf[wave][t & 1][0];

        // charge: v = v*decay + x@Wf.T + (spiked ? WrT[cstar, :] : 0)
        if (spiked) {                                  // wave-uniform branch
            const float* wr = WrT + (size_t)cstar * Csz;
            #pragma unroll
            for (int i = 0; i < 8; ++i) {
                const int c = (i << 5) + lane;
                v[i] = v[i] * DECAY + yrow[c] + wr[c];
            }
        } else {
            #pragma unroll
            for (int i = 0; i < 8; ++i) {
                const int c = (i << 5) + lane;
                v[i] = v[i] * DECAY + yrow[c];
            }
        }

        // WTA argmax over 256 channels (lowest index wins ties, like jnp.argmax)
        float mval = v[0];
        int   midx = lane;
        #pragma unroll
        for (int i = 1; i < 8; ++i) {
            const int c = (i << 5) + lane;
            if (v[i] > mval) { mval = v[i]; midx = c; }
        }
        #pragma unroll
        for (int off = 16; off > 0; off >>= 1) {
            const float om = __shfl_xor(mval, off, 32);
            const int   oi = __shfl_xor(midx, off, 32);
            if (om > mval || (om == mval && oi < midx)) { mval = om; midx = oi; }
        }
        spiked = (mval >= VTH);   // winner spikes iff its v >= VTH
        cstar  = midx;

        // save pre-reset v; hard reset every channel >= VTH (unmasked spike)
        #pragma unroll
        for (int i = 0; i < 8; ++i) {
            pre[i] = v[i];
            if (v[i] >= VTH) v[i] = 0.0f;
        }
    }

    #pragma unroll
    for (int i = 0; i < 8; ++i)
        out[(size_t)b * Csz + (i << 5) + lane] = __expf(pre[i]);
}

// ---------------------------------------------------------------------------
extern "C" void kernel_launch(void* const* d_in, const int* in_sizes, int n_in,
                              void* d_out, int out_size, void* d_ws, size_t ws_size,
                              hipStream_t stream) {
    const float* x  = (const float*)d_in[0];   // [B, C, T]
    const float* Wf = (const float*)d_in[1];   // [C, C]
    const float* Wr = (const float*)d_in[2];   // [C, C]
    float* out = (float*)d_out;                // [B, C]

    const size_t YELEMS = (size_t)Tsz * Bsz * Csz;   // 1 GiB of f32
    float* Yws  = (float*)d_ws;
    float* WrT  = Yws + YELEMS;

    // K0: transpose Wr (one-time, 256 KB)
    wr_transpose_kernel<<<dim3((Csz * Csz) / 256), dim3(256), 0, stream>>>(Wr, WrT);

    // K1: Y = x_t @ Wf.T for all t  (WMMA bf16 -> f32)
    xwf_gemm_kernel<<<dim3(Tsz, Bsz / 16), dim3(256), 0, stream>>>(x, Wf, Yws);

    // K2: sequential WTA-LIF scan + exp (async double-buffered Y stream)
    wta_lif_scan_kernel<<<dim3(Bsz / 8), dim3(256), 0, stream>>>(Yws, WrT, out);
}
